// OneHeadSelfAttentionQKV_80418967650595
// MI455X (gfx1250) — matrix operand order repaired and checked by reading, not compiled, not run on hardware
//
#include <hip/hip_runtime.h>
#include <hip/hip_bf16.h>
#include <stdint.h>

// MI455X / gfx1250, wave32. WMMA bf16 + Tensor Data Mover (TDM) LDS fills.
typedef __attribute__((ext_vector_type(16))) __bf16 v16bf;
typedef __attribute__((ext_vector_type(8)))  float  v8f;
typedef __attribute__((ext_vector_type(4)))  unsigned u32x4;
typedef __attribute__((ext_vector_type(8)))  int      i32x8;
typedef __attribute__((ext_vector_type(4)))  int      i32x4;

#define KDIM 1024
#define LOW  512
#define SEQ  2048

__device__ __forceinline__ unsigned f2bf(float x) {
  union { float f; unsigned u; } c; c.f = x;
  unsigned u = c.u;
  return (u + 0x7FFFu + ((u >> 16) & 1u)) >> 16;   // round-to-nearest-even
}
__device__ __forceinline__ unsigned packbf(float lo, float hi) {
  return f2bf(lo) | (f2bf(hi) << 16);
}

union AB { unsigned u[8]; v16bf v; };

__device__ __forceinline__ v8f wmma_bf16(const AB& a, const AB& b, v8f c) {
  return __builtin_amdgcn_wmma_f32_16x16x32_bf16(false, a.v, false, b.v,
                                                 (short)0, c, false, false);
}
__device__ __forceinline__ v8f zero8() {
  v8f z;
#pragma unroll
  for (int i = 0; i < 8; ++i) z[i] = 0.f;
  return z;
}
__device__ __forceinline__ float rmax16(float x) {
#pragma unroll
  for (int m = 1; m < 16; m <<= 1) x = fmaxf(x, __shfl_xor(x, m, 32));
  return x;
}
__device__ __forceinline__ float rsum16(float x) {
#pragma unroll
  for (int m = 1; m < 16; m <<= 1) x += __shfl_xor(x, m, 32);
  return x;
}

// Low 32 bits of a generic pointer to __shared__ = LDS byte offset.
#define LDS_OFF(p) ((unsigned)(unsigned long long)(uintptr_t)(const void*)(p))

// ---------------------------------------------------------------------------
// TDM: 2D bf16 tile load, global -> LDS. D# per CDNA5 ISA.
// tile_x elems along contiguous dim, tile_y rows, row_stride in elements.
// Group0: [1:0]count=1 | lds_addr | global_addr[56:0] | type=2.
// Group1: data_size=2B; tensor_dim0=row_stride; tensor_dim1=1<<20 (no OOB);
//         tile_dim0/1; tensor_dim0_stride=row_stride; dim1_stride=0 (2D).
// This toolchain (clang-23 / therock-10.0 headers) uses the 6-arg builtin:
//   (u32x4 g0, i32x8 g1, i32x4 g2, i32x4 g3, i32x8 pad, i32 cpol)
// ---------------------------------------------------------------------------
__device__ __forceinline__ void tdm_load_2d(unsigned lds_off, const void* gsrc,
                                            unsigned tile_x, unsigned tile_y,
                                            unsigned row_stride) {
  unsigned long long ga = (unsigned long long)(uintptr_t)gsrc;
  u32x4 g0;
  g0[0] = 1u;                                                 // count=1, user
  g0[1] = lds_off;                                            // lds_addr
  g0[2] = (unsigned)ga;                                       // gaddr[31:0]
  g0[3] = (unsigned)((ga >> 32) & 0x01ffffffu) | (2u << 30);  // gaddr[56:32]|type=2
  i32x8 g1;
  g1[0] = 0x00010000;                            // wg_mask=0, data_size=2B
  g1[1] = (int)((row_stride & 0xffffu) << 16);   // tensor_dim0[15:0] @ [63:48]
  g1[2] = (int)((row_stride >> 16) & 0xffffu);   // tensor_dim0[31:16]; dim1 lo=0
  g1[3] = (int)(16u | (tile_x << 16));           // tensor_dim1=1<<20 hi; tile_dim0
  g1[4] = (int)tile_y;                           // tile_dim1; tile_dim2=0 (2D)
  g1[5] = (int)row_stride;                       // tensor_dim0_stride[31:0]
  g1[6] = 0;                                     // stride0 hi; stride1 lo
  g1[7] = 0;                                     // stride1 hi
  i32x4 z4 = {0, 0, 0, 0};
  i32x8 z8;
#pragma unroll
  for (int i = 0; i < 8; ++i) z8[i] = 0;
  __builtin_amdgcn_tensor_load_to_lds(g0, g1, z4, z4, z8, 0);
}

// ---------------------------------------------------------------------------
// Kernel 1: combined weights  Wc = Wx @ Wr   (fp32 math, bf16 output)
// ---------------------------------------------------------------------------
__global__ void combine_w_kernel(const float* __restrict__ Wq,
                                 const float* __restrict__ Wk,
                                 const float* __restrict__ Wv,
                                 const float* __restrict__ Wr,
                                 unsigned short* __restrict__ wc_out) {
  const float* Wx = (blockIdx.y == 0) ? Wq : (blockIdx.y == 1) ? Wk : Wv;
  int idx = blockIdx.x * 256 + threadIdx.x;   // 0 .. 512*1024-1
  int n = idx >> 10, k = idx & 1023;
  const float* wrow = Wx + n * LOW;
  float acc = 0.f;
#pragma unroll 8
  for (int d = 0; d < LOW; ++d) acc += wrow[d] * Wr[d * KDIM + k];
  wc_out[(size_t)blockIdx.y * (LOW * KDIM) + idx] = (unsigned short)f2bf(acc);
}

// ---------------------------------------------------------------------------
// Kernel 2: projection GEMM  Y[16384,512] = X[16384,1024] @ Wc^T
// X fp32 -> bf16 through VGPRs; W tiles via TDM, double-buffered.
// ---------------------------------------------------------------------------
__global__ void proj_kernel(const float* __restrict__ Q,
                            const float* __restrict__ K,
                            const float* __restrict__ V,
                            const unsigned short* __restrict__ wc,
                            unsigned short* __restrict__ qb,
                            unsigned short* __restrict__ kb,
                            unsigned short* __restrict__ vb) {
  __shared__ unsigned sX[128 * 16];        // 128 rows x 32 bf16
  __shared__ unsigned sW[2][128 * 16];     // double-buffered TDM dest

  const float* X = (blockIdx.z == 0) ? Q : (blockIdx.z == 1) ? K : V;
  unsigned short* Y = (blockIdx.z == 0) ? qb : (blockIdx.z == 1) ? kb : vb;
  const unsigned short* Wbase =
      wc + (size_t)blockIdx.z * (LOW * KDIM) + (size_t)blockIdx.y * 128 * KDIM;

  const int t = threadIdx.x, lane = t & 31, wave = t >> 5;
  const int half = lane >> 4, ln = lane & 15;
  const int wm = wave >> 1, wn = wave & 1;
  const int M0 = blockIdx.x * 128, N0 = blockIdx.y * 128;
  const int lrow = t >> 1, lh = t & 1;
  const unsigned ldsW[2] = {LDS_OFF(&sW[0][0]), LDS_OFF(&sW[1][0])};

  v8f acc[2][4];
#pragma unroll
  for (int mt = 0; mt < 2; ++mt)
#pragma unroll
    for (int nt = 0; nt < 4; ++nt) acc[mt][nt] = zero8();

  if (wave == 0)                                   // prologue: W chunk 0
    tdm_load_2d(ldsW[0], Wbase, 32, 128, KDIM);

  for (int kc = 0; kc < 32; ++kc) {
    __syncthreads();                               // prior tile consumers done
    {  // X tile: each thread 16 fp32 -> 8 packed bf16 pairs
      const float4* Xv = (const float4*)X;
      size_t xb = (size_t)(M0 + lrow) * 256 + kc * 8 + lh * 4;
#pragma unroll
      for (int i = 0; i < 4; ++i) {
        float4 f = Xv[xb + i];
        sX[lrow * 16 + lh * 8 + i * 2]     = packbf(f.x, f.y);
        sX[lrow * 16 + lh * 8 + i * 2 + 1] = packbf(f.z, f.w);
      }
    }
    if (wave == 0) {                               // stream next W, wait cur
      if (kc + 1 < 32) {
        tdm_load_2d(ldsW[(kc + 1) & 1], Wbase + (kc + 1) * 32, 32, 128, KDIM);
        __builtin_amdgcn_s_wait_tensorcnt(1);
      } else {
        __builtin_amdgcn_s_wait_tensorcnt(0);
      }
    }
    __syncthreads();

    const unsigned* sWc = sW[kc & 1];
    AB a[2], bfr[4];
#pragma unroll
    for (int mt = 0; mt < 2; ++mt)
#pragma unroll
      for (int j = 0; j < 8; ++j)     // A 16x32: K = 2j + 16*half
        a[mt].u[j] = sX[(wm * 32 + mt * 16 + ln) * 16 + j + 8 * half];
#pragma unroll
    for (int nt = 0; nt < 4; ++nt)
#pragma unroll
      for (int j = 0; j < 8; ++j)     // B 32x16: K = 2j + 16*half
        bfr[nt].u[j] = sWc[(wn * 64 + nt * 16 + ln) * 16 + j + 8 * half];
#pragma unroll
    for (int mt = 0; mt < 2; ++mt)
#pragma unroll
      for (int nt = 0; nt < 4; ++nt)
        acc[mt][nt] = wmma_bf16(a[mt], bfr[nt], acc[mt][nt]);
  }

  // C/D layout: row = r + 8*half, col = ln
#pragma unroll
  for (int mt = 0; mt < 2; ++mt)
#pragma unroll
    for (int nt = 0; nt < 4; ++nt)
#pragma unroll
      for (int r = 0; r < 8; ++r) {
        int row = M0 + wm * 32 + mt * 16 + r + 8 * half;
        int col = N0 + wn * 64 + nt * 16 + ln;
        Y[(size_t)row * LOW + col] = (unsigned short)f2bf(acc[mt][nt][r]);
      }
}

// ---------------------------------------------------------------------------
// Kernel 3: flash attention, TDM-fed. Block = 32 query rows of one batch.
// 8 waves: wr (16-row subtile), wc4 (16-key S column / 128-col O,V group).
// Q once via TDM; K chunks double-buffered TDM; full 64x512 V block TDM
// issued before softmax bookkeeping (DMA overlaps exp/reductions).
// ---------------------------------------------------------------------------
__global__ void attn_kernel(const unsigned short* __restrict__ qb,
                            const unsigned short* __restrict__ kb,
                            const unsigned short* __restrict__ vb,
                            float* __restrict__ out) {
  __shared__ unsigned sQ[32 * 256];          // 32 x 512 bf16 (32 KB)
  __shared__ unsigned short sV[64 * 512];    // 64 keys x 512 (64 KB, TDM dest)
  __shared__ unsigned sK[2][64 * 16];        // 2 x (64 keys x 32 feats) (8 KB)
  __shared__ unsigned sP[32 * 32];           // P: 32 x 64 bf16 (4 KB)
  __shared__ float sRowM[32], sRowL[32], sMnew[32], sAlpha[32], sRed[32 * 4];

  const int t = threadIdx.x, lane = t & 31, wave = t >> 5;
  const int half = lane >> 4, ln = lane & 15;
  const int wr = wave >> 2, wc4 = wave & 3;
  const int b = blockIdx.y, R0 = blockIdx.x * 32;
  const float SCALE = 0.04419417382415922f;  // 1/sqrt(512)

  const unsigned ldsQ = LDS_OFF(sQ);
  const unsigned ldsV = LDS_OFF(sV);
  const unsigned ldsK[2] = {LDS_OFF(&sK[0][0]), LDS_OFF(&sK[1][0])};

  if (t < 32) { sRowM[t] = -1e30f; sRowL[t] = 0.f; }
  if (wave == 0)  // Q tile once (drained together with first K chunk)
    tdm_load_2d(ldsQ, qb + (size_t)(b * SEQ + R0) * LOW, 512, 32, LOW);

  v8f o[8];
#pragma unroll
  for (int nt = 0; nt < 8; ++nt) o[nt] = zero8();

  for (int jb = 0; jb < 32; ++jb) {
    const int j0 = jb * 64;
    const unsigned short* kblk = kb + (size_t)(b * SEQ + j0) * LOW;

    if (wave == 0)                                  // K chunk 0 for this block
      tdm_load_2d(ldsK[0], kblk, 32, 64, LOW);

    // ----- S = q @ k^T for this 64-key block (wave: 16 rows x 16 keys) -----
    v8f s = zero8();
    for (int kc = 0; kc < 16; ++kc) {
      if (wave == 0) {                              // stream next, wait current
        if (kc + 1 < 16) {
          tdm_load_2d(ldsK[(kc + 1) & 1], kblk + (kc + 1) * 32, 32, 64, LOW);
          __builtin_amdgcn_s_wait_tensorcnt(1);
        } else {
          __builtin_amdgcn_s_wait_tensorcnt(0);
        }
      }
      __syncthreads();                              // chunk kc visible to all
      const unsigned* sKc = sK[kc & 1];
      AB a, bk;
#pragma unroll
      for (int j = 0; j < 8; ++j)
        a.u[j] = sQ[(wr * 16 + ln) * 256 + kc * 16 + j + 8 * half];
#pragma unroll
      for (int j = 0; j < 8; ++j)  // B[feat,key] = k[key][feat]; pairs contiguous
        bk.u[j] = sKc[(wc4 * 16 + ln) * 16 + j + 8 * half];
      s = wmma_bf16(a, bk, s);
      __syncthreads();                              // readers done before reissue
    }

    if (wave == 0)  // launch V block DMA now; overlaps softmax bookkeeping
      tdm_load_2d(ldsV, vb + (size_t)(b * SEQ + j0) * LOW, 512, 64, LOW);

    // ----- online softmax: block row-max (16-lane butterfly + LDS combine) --
#pragma unroll
    for (int r = 0; r < 8; ++r) {
      float x = s[r] * SCALE;
      s[r] = x;
      float mx = rmax16(x);
      if (ln == 0) sRed[(wr * 16 + r + 8 * half) * 4 + wc4] = mx;
    }
    __syncthreads();
    if (t < 32) {
      float mb = fmaxf(fmaxf(sRed[t * 4], sRed[t * 4 + 1]),
                       fmaxf(sRed[t * 4 + 2], sRed[t * 4 + 3]));
      float mo = sRowM[t];
      float mn = fmaxf(mo, mb);
      sMnew[t] = mn;
      sAlpha[t] = __expf(mo - mn);
      sRowM[t] = mn;
    }
    __syncthreads();

    // ----- P = exp(S - m_new), row sums, rescale O accumulators -----
    unsigned short* sPh = (unsigned short*)sP;
#pragma unroll
    for (int r = 0; r < 8; ++r) {
      int row = wr * 16 + r + 8 * half;
      float p = __expf(s[r] - sMnew[row]);
      sPh[row * 64 + wc4 * 16 + ln] = (unsigned short)f2bf(p);
      float ps = rsum16(p);
      if (ln == 0) sRed[row * 4 + wc4] = ps;
      float al = sAlpha[row];
#pragma unroll
      for (int nt = 0; nt < 8; ++nt) o[nt][r] *= al;
    }
    __syncthreads();
    if (t < 32)
      sRowL[t] = sRowL[t] * sAlpha[t] +
                 (sRed[t * 4] + sRed[t * 4 + 1] + sRed[t * 4 + 2] + sRed[t * 4 + 3]);

    if (wave == 0) __builtin_amdgcn_s_wait_tensorcnt(0);  // V block landed
    __syncthreads();

    // ----- O += P @ V (wave: 16 rows x 128 cols; no barriers between halves)
#pragma unroll
    for (int ks = 0; ks < 2; ++ks) {
      AB ap;
#pragma unroll
      for (int j = 0; j < 8; ++j)  // A = P rows, K = key dim (pairs contiguous)
        ap.u[j] = sP[(wr * 16 + ln) * 32 + ks * 16 + j + 8 * half];
#pragma unroll
      for (int nt = 0; nt < 8; ++nt) {
        AB bv;
        int col = wc4 * 128 + nt * 16 + ln;
#pragma unroll
        for (int j = 0; j < 8; ++j) {  // B pairs along key dim: two b16 gathers
          int kk = ks * 32 + 2 * j + 16 * half;
          bv.u[j] = (unsigned)sV[kk * 512 + col] |
                    ((unsigned)sV[(kk + 1) * 512 + col] << 16);
        }
        o[nt] = wmma_bf16(ap, bv, o[nt]);
      }
    }
  }

  __syncthreads();
  float linv[8];
#pragma unroll
  for (int r = 0; r < 8; ++r) linv[r] = 1.0f / sRowL[wr * 16 + r + 8 * half];
#pragma unroll
  for (int nt = 0; nt < 8; ++nt)
#pragma unroll
    for (int r = 0; r < 8; ++r) {
      int row = R0 + wr * 16 + r + 8 * half;
      int col = wc4 * 128 + nt * 16 + ln;
      out[((size_t)b * SEQ + row) * LOW + col] = o[nt][r] * linv[r];
    }
}

// ---------------------------------------------------------------------------
extern "C" void kernel_launch(void* const* d_in, const int* in_sizes, int n_in,
                              void* d_out, int out_size, void* d_ws, size_t ws_size,
                              hipStream_t stream) {
  (void)in_sizes; (void)n_in; (void)out_size; (void)ws_size;
  const float* Q  = (const float*)d_in[0];
  const float* K  = (const float*)d_in[1];
  const float* V  = (const float*)d_in[2];
  const float* Wr = (const float*)d_in[3];
  const float* Wq = (const float*)d_in[4];
  const float* Wk = (const float*)d_in[5];
  const float* Wv = (const float*)d_in[6];

  unsigned short* wc = (unsigned short*)d_ws;              // 3 * 512*1024
  unsigned short* qb = wc + (size_t)3 * LOW * KDIM;        // 8*2048*512 each
  unsigned short* kb = qb + (size_t)8 * SEQ * LOW;
  unsigned short* vb = kb + (size_t)8 * SEQ * LOW;

  combine_w_kernel<<<dim3(2048, 3), 256, 0, stream>>>(Wq, Wk, Wv, Wr, wc);
  proj_kernel<<<dim3(128, 4, 3), 256, 0, stream>>>(Q, K, V, wc, qb, kb, vb);
  attn_kernel<<<dim3(64, 8), 256, 0, stream>>>(qb, kb, vb, (float*)d_out);
}
